// SelfAttention_41068477285217
// MI455X (gfx1250) — compile-verified, matrix operand order checked by reading, not run on hardware
//
#include <hip/hip_runtime.h>
#include <hip/hip_bf16.h>
#include <math.h>
#include <stdint.h>

typedef __attribute__((ext_vector_type(16))) __bf16 v16bf;
typedef __attribute__((ext_vector_type(8)))  __bf16 v8bf;
typedef __attribute__((ext_vector_type(8)))  float  v8f;
typedef __attribute__((ext_vector_type(4)))  unsigned int v4u;
typedef __attribute__((ext_vector_type(8)))  int v8i;
typedef __attribute__((ext_vector_type(4)))  int v4i;

static constexpr int BATCH = 2;
static constexpr int SEQ   = 2048;
static constexpr int EMB   = 1024;
static constexpr int HEADS = 16;
static constexpr int DKH   = 64;
static constexpr int MROWS = BATCH * SEQ;   // 4096

#define BM 128
#define BN 64
#define BK 32

__device__ __forceinline__ __bf16 f2bf(float f) { return (__bf16)f; }

__device__ __forceinline__ v16bf cat8(v8bf lo, v8bf hi) {
  return __builtin_shufflevector(lo, hi, 0,1,2,3,4,5,6,7,8,9,10,11,12,13,14,15);
}

__device__ __forceinline__ v8f wmma_bf16(v16bf a, v16bf b, v8f c) {
  // D = A(16x32 bf16) * B(32x16 bf16) + C(16x16 f32)
  return __builtin_amdgcn_wmma_f32_16x16x32_bf16(
      /*neg_a=*/false, a, /*neg_b=*/false, b,
      /*c_mod=*/(short)0, c, /*reuse_a=*/false, /*reuse_b=*/false);
}

// ---------------------------------------------------------------------------
// Kernel 1: QKV projection.  C(4096 x 3072) = X(4096x1024) * W_in^T + b_in
// Scatters q (scaled by 1/sqrt(DK)) -> (B,H,S,DK), k -> (B,H,S,DK),
// v -> (B,H,DK,S)  [transposed for contiguous PV B-fragment loads], all bf16.
// ---------------------------------------------------------------------------
__global__ __launch_bounds__(256) void qkv_gemm_wmma(
    const float* __restrict__ x, const float* __restrict__ w_in,
    const float* __restrict__ b_in,
    __bf16* __restrict__ qbuf, __bf16* __restrict__ kbuf,
    __bf16* __restrict__ vbuf)
{
  __shared__ __align__(16) __bf16 As[BM][BK];   // 8 KB
  __shared__ __align__(16) __bf16 Bs[BN][BK];   // 4 KB
  const int tid  = threadIdx.x;
  const int wave = tid >> 5;
  const int lane = tid & 31;
  const int m0 = blockIdx.x * BM;
  const int n0 = blockIdx.y * BN;
  const int wm = (wave >> 1) * 32;      // 4 waves along M
  const int wn = (wave & 1) * 32;       // 2 waves along N
  const int ln16 = lane & 15;
  const int half = lane >> 4;
  const int kb   = half * 8;

  v8f acc[2][2] = {};

  const int ar = tid >> 1;            // A staging: row 0..127
  const int ac = (tid & 1) * 16;      // 16 floats each
  const int br = tid >> 2;            // B staging: row 0..63
  const int bc = (tid & 3) * 8;       // 8 floats each

  for (int k0 = 0; k0 < EMB; k0 += BK) {
    {
      const float4* src = (const float4*)(x + (size_t)(m0 + ar) * EMB + k0 + ac);
      #pragma unroll
      for (int i = 0; i < 4; ++i) {
        float4 v = src[i];
        As[ar][ac + 4*i + 0] = f2bf(v.x);
        As[ar][ac + 4*i + 1] = f2bf(v.y);
        As[ar][ac + 4*i + 2] = f2bf(v.z);
        As[ar][ac + 4*i + 3] = f2bf(v.w);
      }
    }
    {
      const float4* src = (const float4*)(w_in + (size_t)(n0 + br) * EMB + k0 + bc);
      #pragma unroll
      for (int i = 0; i < 2; ++i) {
        float4 v = src[i];
        Bs[br][bc + 4*i + 0] = f2bf(v.x);
        Bs[br][bc + 4*i + 1] = f2bf(v.y);
        Bs[br][bc + 4*i + 2] = f2bf(v.z);
        Bs[br][bc + 4*i + 3] = f2bf(v.w);
      }
    }
    __syncthreads();
    v16bf afrag[2], bfrag[2];
    #pragma unroll
    for (int i = 0; i < 2; ++i) {
      const __bf16* ap = &As[wm + i*16 + ln16][0];
      afrag[i] = cat8(*(const v8bf*)(ap + kb), *(const v8bf*)(ap + kb + 16));
      const __bf16* bp = &Bs[wn + i*16 + ln16][0];
      bfrag[i] = cat8(*(const v8bf*)(bp + kb), *(const v8bf*)(bp + kb + 16));
    }
    #pragma unroll
    for (int i = 0; i < 2; ++i)
      #pragma unroll
      for (int j = 0; j < 2; ++j)
        acc[i][j] = wmma_bf16(afrag[i], bfrag[j], acc[i][j]);
    __syncthreads();
  }

  // Epilogue: bias + scatter.  which = q/k/v select is BLOCK-UNIFORM
  // (n0 is a multiple of 64 and 64 | 1024), so this is one scalar branch.
  const int which = n0 >> 10;
  #pragma unroll
  for (int j = 0; j < 2; ++j) {
    const int n = n0 + wn + j*16 + ln16;
    const int e = n & (EMB - 1);
    const int hh = e >> 6;
    const int d = e & (DKH - 1);
    const float bias = b_in[n];
    #pragma unroll
    for (int i = 0; i < 2; ++i) {
      #pragma unroll
      for (int r = 0; r < 8; ++r) {
        const int m = m0 + wm + i*16 + half*8 + r;
        const int bidx = m >> 11;
        const int s = m & (SEQ - 1);
        const float val = acc[i][j][r] + bias;
        const size_t bh = (size_t)bidx * HEADS + hh;
        if (which == 0)
          qbuf[(bh * SEQ + s) * DKH + d] = f2bf(val * 0.125f);   // 1/sqrt(64)
        else if (which == 1)
          kbuf[(bh * SEQ + s) * DKH + d] = f2bf(val);
        else
          vbuf[(bh * DKH + d) * SEQ + s] = f2bf(val);            // transposed
      }
    }
  }
}

// ---------------------------------------------------------------------------
// Kernel 2: causal flash attention.  grid = (B*H, S/128), 8 waves/block,
// each wave owns 16 query rows, streams key tiles of 32 with online softmax.
// ---------------------------------------------------------------------------
__global__ __launch_bounds__(256) void flash_attn_wmma(
    const __bf16* __restrict__ qbuf, const __bf16* __restrict__ kbuf,
    const __bf16* __restrict__ vbuf, __bf16* __restrict__ obuf)
{
  __shared__ __align__(16) __bf16 pbuf[8][16][32];   // per-wave P transpose tile
  const int bh   = blockIdx.x;
  const int bidx = bh >> 4;
  const int h    = bh & 15;
  const int tid  = threadIdx.x;
  const int wave = tid >> 5;
  const int lane = tid & 31;
  const int q0   = blockIdx.y * 128 + wave * 16;
  const __bf16* Q = qbuf + (size_t)bh * SEQ * DKH;
  const __bf16* K = kbuf + (size_t)bh * SEQ * DKH;
  const __bf16* V = vbuf + (size_t)bh * DKH * SEQ;   // (DK, S)

  const int col  = lane & 15;
  const int half = lane >> 4;
  const int kb   = half * 8;

  // Q fragments (16 rows x 64 dk) stay in registers for the whole loop.
  v16bf qa[2];
  #pragma unroll
  for (int ks = 0; ks < 2; ++ks) {
    const __bf16* qp = Q + (size_t)(q0 + col) * DKH + ks*32;
    qa[ks] = cat8(*(const v8bf*)(qp + kb), *(const v8bf*)(qp + kb + 16));
  }

  v8f o[4] = {};
  float mrow[8], lrow[8];
  #pragma unroll
  for (int r = 0; r < 8; ++r) { mrow[r] = -__builtin_inff(); lrow[r] = 0.f; }

  for (int kt = 0; kt <= q0 + 15; kt += 32) {
    // Prefetch next key/value tiles into cache while this tile computes.
    if (kt + 32 <= q0 + 15) {
      __builtin_prefetch(K + (size_t)(kt + 32 + col) * DKH, 0, 0);
      __builtin_prefetch(K + (size_t)(kt + 48 + col) * DKH, 0, 0);
      __builtin_prefetch(V + (size_t)col * SEQ + kt + 32, 0, 0);
    }
    // S = Q * K^T for 32 keys (two 16-key column blocks, 2 k-steps over DK).
    v8f sc[2] = {};
    #pragma unroll
    for (int blk = 0; blk < 2; ++blk) {
      #pragma unroll
      for (int ks = 0; ks < 2; ++ks) {
        const __bf16* kp = K + (size_t)(kt + blk*16 + col) * DKH + ks*32;
        v16bf kfrag = cat8(*(const v8bf*)(kp + kb), *(const v8bf*)(kp + kb + 16));
        sc[blk] = wmma_bf16(qa[ks], kfrag, sc[blk]);
      }
    }
    // Causal mask (C layout: row = half*8+r, col = kt + blk*16 + col).
    #pragma unroll
    for (int r = 0; r < 8; ++r) {
      const int qrow = q0 + half*8 + r;
      if (kt + col > qrow)      sc[0][r] = -__builtin_inff();
      if (kt + 16 + col > qrow) sc[1][r] = -__builtin_inff();
    }
    // Online softmax: row reductions are butterflies over the 16-lane half.
    float tmax[8];
    #pragma unroll
    for (int r = 0; r < 8; ++r) tmax[r] = fmaxf(sc[0][r], sc[1][r]);
    #pragma unroll
    for (int msk = 1; msk < 16; msk <<= 1)
      #pragma unroll
      for (int r = 0; r < 8; ++r) tmax[r] = fmaxf(tmax[r], __shfl_xor(tmax[r], msk, 32));
    float p0[8], p1[8], tsum[8], scl[8];
    #pragma unroll
    for (int r = 0; r < 8; ++r) {
      const float mn = fmaxf(mrow[r], tmax[r]);
      scl[r]  = __expf(mrow[r] - mn);
      mrow[r] = mn;
      p0[r] = __expf(sc[0][r] - mn);
      p1[r] = __expf(sc[1][r] - mn);
      tsum[r] = p0[r] + p1[r];
    }
    #pragma unroll
    for (int msk = 1; msk < 16; msk <<= 1)
      #pragma unroll
      for (int r = 0; r < 8; ++r) tsum[r] += __shfl_xor(tsum[r], msk, 32);
    #pragma unroll
    for (int r = 0; r < 8; ++r) lrow[r] = lrow[r] * scl[r] + tsum[r];
    #pragma unroll
    for (int c = 0; c < 4; ++c)
      #pragma unroll
      for (int r = 0; r < 8; ++r) o[c][r] *= scl[r];
    // Transpose P from C layout to A layout through per-wave LDS tile.
    #pragma unroll
    for (int r = 0; r < 8; ++r) {
      pbuf[wave][half*8 + r][col]      = f2bf(p0[r]);
      pbuf[wave][half*8 + r][16 + col] = f2bf(p1[r]);
    }
    const __bf16* pp = &pbuf[wave][col][0];
    v16bf pa = cat8(*(const v8bf*)(pp + kb), *(const v8bf*)(pp + kb + 16));
    // O += P * V  (V transposed: B[k][n] = V[kt+k][c*16+n] = Vt[c*16+n][kt+k]).
    #pragma unroll
    for (int c = 0; c < 4; ++c) {
      const __bf16* vp = V + (size_t)(c*16 + col) * SEQ + kt;
      v16bf vfrag = cat8(*(const v8bf*)(vp + kb), *(const v8bf*)(vp + kb + 16));
      o[c] = wmma_bf16(pa, vfrag, o[c]);
    }
  }
  // Finalize: O /= l, write bf16 (B,S,E) for the output projection.
  #pragma unroll
  for (int c = 0; c < 4; ++c) {
    #pragma unroll
    for (int r = 0; r < 8; ++r) {
      const int qrow = q0 + half*8 + r;
      const float val = o[c][r] / lrow[r];
      obuf[((size_t)bidx * SEQ + qrow) * EMB + h*DKH + c*16 + col] = f2bf(val);
    }
  }
}

// ---------------------------------------------------------------------------
// Kernel 3: output projection.  Y(4096x1024) = A_bf16 * W_out^T + b_out (f32)
// The bf16 A tile (128x32) is DMAed into LDS by the Tensor Data Mover:
// one TENSOR_LOAD_TO_LDS per K-step, issued by wave 0, tracked by TENSORcnt.
// ---------------------------------------------------------------------------
__global__ __launch_bounds__(256) void out_gemm_wmma(
    const __bf16* __restrict__ ain, const float* __restrict__ w_out,
    const float* __restrict__ b_out, float* __restrict__ y)
{
  __shared__ __align__(16) __bf16 As[BM][BK];
  __shared__ __align__(16) __bf16 Bs[BN][BK];
  const int tid  = threadIdx.x;
  const int wave = tid >> 5;
  const int lane = tid & 31;
  const int m0 = blockIdx.x * BM;
  const int n0 = blockIdx.y * BN;
  const int wm = (wave >> 1) * 32;
  const int wn = (wave & 1) * 32;
  const int ln16 = lane & 15;
  const int half = lane >> 4;
  const int kb   = half * 8;

  v8f acc[2][2] = {};

  const int br = tid >> 2;
  const int bc = (tid & 3) * 8;

  // Generic pointer to LDS: low 32 bits are the LDS byte offset.
  const unsigned lds_as = (unsigned)(uintptr_t)&As[0][0];
  // Byte address of this block's A tile rows (column offset added per K-step).
  const uint64_t abase = (uint64_t)(uintptr_t)ain + (uint64_t)m0 * EMB * 2u;

  for (int k0 = 0; k0 < EMB; k0 += BK) {
    if (wave == 0) {
      // --- Tensor Data Mover: DMA 128x32 bf16 tile -> LDS (row-major) ---
      const uint64_t ga = abase + (uint64_t)k0 * 2u;
      v4u g0 = { 1u,                                   // count=1, user mode
                 lds_as,                               // lds_addr
                 (unsigned)ga,                         // global_addr[31:0]
                 ((unsigned)(ga >> 32) & 0x1FFFFFFu) | (2u << 30) }; // +type=2
      v8i g1 = { (int)(1u << 16),                      // data_size = 2 bytes
                 (int)(1024u << 16),                   // tensor_dim0 = 1024
                 (int)(4096u << 16),                   // tensor_dim1 = 4096
                 (int)(32u << 16),                     // tile_dim0 = 32
                 (int)128,                             // tile_dim1 = 128
                 (int)1024,                            // tensor_dim0_stride
                 0, 0 };
      v4i g2 = { 0, 0, 0, 0 };
      v4i g3 = { 0, 0, 0, 0 };
      v8i g4 = { 0, 0, 0, 0, 0, 0, 0, 0 };   // unused higher-dim group (2D tensor)
      __builtin_amdgcn_tensor_load_to_lds(g0, g1, g2, g3, g4, 0);
    }
    // B tile staged by all waves (fp32 -> bf16 conversion needed).
    {
      const float4* src = (const float4*)(w_out + (size_t)(n0 + br) * EMB + k0 + bc);
      #pragma unroll
      for (int i = 0; i < 2; ++i) {
        float4 v = src[i];
        Bs[br][bc + 4*i + 0] = f2bf(v.x);
        Bs[br][bc + 4*i + 1] = f2bf(v.y);
        Bs[br][bc + 4*i + 2] = f2bf(v.z);
        Bs[br][bc + 4*i + 3] = f2bf(v.w);
      }
    }
    if (wave == 0) __builtin_amdgcn_s_wait_tensorcnt((short)0);
    __syncthreads();
    v16bf afrag[2], bfrag[2];
    #pragma unroll
    for (int i = 0; i < 2; ++i) {
      const __bf16* ap = &As[wm + i*16 + ln16][0];
      afrag[i] = cat8(*(const v8bf*)(ap + kb), *(const v8bf*)(ap + kb + 16));
      const __bf16* bp = &Bs[wn + i*16 + ln16][0];
      bfrag[i] = cat8(*(const v8bf*)(bp + kb), *(const v8bf*)(bp + kb + 16));
    }
    #pragma unroll
    for (int i = 0; i < 2; ++i)
      #pragma unroll
      for (int j = 0; j < 2; ++j)
        acc[i][j] = wmma_bf16(afrag[i], bfrag[j], acc[i][j]);
    __syncthreads();   // protects As (next TDM write) and Bs (next stage)
  }

  #pragma unroll
  for (int j = 0; j < 2; ++j) {
    const int n = n0 + wn + j*16 + ln16;
    const float bias = b_out[n];
    #pragma unroll
    for (int i = 0; i < 2; ++i) {
      #pragma unroll
      for (int r = 0; r < 8; ++r) {
        const int m = m0 + wm + i*16 + half*8 + r;
        y[(size_t)m * EMB + n] = acc[i][j][r] + bias;
      }
    }
  }
}

// ---------------------------------------------------------------------------
extern "C" void kernel_launch(void* const* d_in, const int* in_sizes, int n_in,
                              void* d_out, int out_size, void* d_ws, size_t ws_size,
                              hipStream_t stream) {
  (void)in_sizes; (void)n_in; (void)out_size; (void)ws_size;
  const float* x     = (const float*)d_in[0];
  const float* w_in  = (const float*)d_in[1];
  const float* b_in  = (const float*)d_in[2];
  const float* w_out = (const float*)d_in[3];
  const float* b_out = (const float*)d_in[4];
  float* out = (float*)d_out;

  // Workspace: q, k, v, attn-out (each 4096*1024 bf16 = 8 MB; 32 MB total).
  const size_t elems = (size_t)MROWS * EMB;
  __bf16* qbuf = (__bf16*)d_ws;
  __bf16* kbuf = qbuf + elems;
  __bf16* vbuf = kbuf + elems;
  __bf16* obuf = vbuf + elems;

  dim3 g1(MROWS / BM, (3 * EMB) / BN);       // 32 x 48
  qkv_gemm_wmma<<<g1, 256, 0, stream>>>(x, w_in, b_in, qbuf, kbuf, vbuf);

  dim3 g2(BATCH * HEADS, SEQ / 128);          // 32 x 16
  flash_attn_wmma<<<g2, 256, 0, stream>>>(qbuf, kbuf, vbuf, obuf);

  dim3 g3(MROWS / BM, EMB / BN);              // 32 x 16
  out_gemm_wmma<<<g3, 256, 0, stream>>>(obuf, w_out, b_out, out);
}
